// TrainableKMeans_10883447128775
// MI455X (gfx1250) — matrix operand order repaired and checked by reading, compiled, not run on hardware
//
#include <hip/hip_runtime.h>

typedef __attribute__((ext_vector_type(16))) _Float16 v16h;
typedef __attribute__((ext_vector_type(8)))  _Float16 v8h;
typedef __attribute__((ext_vector_type(8)))  float    v8f;

#define WAVES 4
#define TPB   (WAVES * 32)
#define D_    64
#define C_    256
#define F_    8
#define LOG2E 1.4426950408889634f

__global__ __launch_bounds__(TPB) void tkmeans_wmma_kernel(
    const float* __restrict__ x,      // [T, 8] last col = category id
    const float* __restrict__ We_g,   // [7, 64]
    const float* __restrict__ be_g,   // [64]
    const float* __restrict__ emb,    // [366, 64]
    const float* __restrict__ cent_g, // [256, 64]
    const float* __restrict__ W2_g,   // [64, 1]
    const float* __restrict__ b2_g,   // [1]
    float* __restrict__ out)          // [T]
{
    // --- LDS ---
    __shared__ __align__(32) _Float16 cent [C_][D_];   // B for distances (K = d)
    __shared__ __align__(32) _Float16 centT[D_][C_];   // B for combine   (K = c)
    __shared__ float c2s[C_];
    __shared__ float We[7 * D_];
    __shared__ float be[D_];
    __shared__ float w2v[D_];
    __shared__ float b2s;
    __shared__ __align__(16) float     xbuf [WAVES][16][F_];
    __shared__ __align__(16) float     hbuf [WAVES][16 * D_];
    __shared__ __align__(16) float     h2buf[WAVES][16];
    __shared__ __align__(32) _Float16  wbuf [WAVES][16 * C_];  // unnormalized softmax numerators

    const int tid  = threadIdx.x;
    const int wv   = tid >> 5;
    const int lane = tid & 31;
    const int row  = lane & 15;   // M (or N) index within 16
    const int hi   = lane >> 4;   // half-wave select

    // ---------------- block-wide preload ----------------
    for (int r = tid; r < C_; r += TPB) {
        const float4* cr = reinterpret_cast<const float4*>(cent_g + r * D_);
        float s = 0.f;
        #pragma unroll
        for (int q = 0; q < 16; ++q) {
            float4 v = cr[q];
            s += v.x * v.x + v.y * v.y + v.z * v.z + v.w * v.w;
            const int d = q * 4;
            cent[r][d + 0] = (_Float16)v.x;  cent[r][d + 1] = (_Float16)v.y;
            cent[r][d + 2] = (_Float16)v.z;  cent[r][d + 3] = (_Float16)v.w;
            centT[d + 0][r] = (_Float16)v.x; centT[d + 1][r] = (_Float16)v.y;
            centT[d + 2][r] = (_Float16)v.z; centT[d + 3][r] = (_Float16)v.w;
        }
        c2s[r] = s;
    }
    for (int i = tid; i < 7 * D_; i += TPB) We[i] = We_g[i];
    if (tid < D_) { be[tid] = be_g[tid]; w2v[tid] = W2_g[tid]; }
    if (tid == 0) b2s = b2_g[0];
    __syncthreads();

    // ---------------- per-wave 16-token tile ----------------
    const long tile  = (long)blockIdx.x * WAVES + wv;
    const long tbase = tile * 16;

    // stage x rows (16 tokens x 8 features)
    if (lane < 16) {
        const float4* xr = reinterpret_cast<const float4*>(x + (tbase + lane) * F_);
        *reinterpret_cast<float4*>(&xbuf[wv][lane][0]) = xr[0];
        *reinterpret_cast<float4*>(&xbuf[wv][lane][4]) = xr[1];
    }
    __syncthreads();

    // h = x[:, :7] @ W_embed + b + emb_table[idx]; each lane owns columns d0, d0+1
    {
        const int d0 = lane * 2;
        float wc0[7], wc1[7];
        #pragma unroll
        for (int f = 0; f < 7; ++f) { wc0[f] = We[f * D_ + d0]; wc1[f] = We[f * D_ + d0 + 1]; }
        const float b0 = be[d0], b1 = be[d0 + 1];
        #pragma unroll
        for (int m = 0; m < 16; ++m) {
            const float* xr = xbuf[wv][m];
            const int idx = (int)xr[7];
            const float2 ev = *reinterpret_cast<const float2*>(emb + (size_t)idx * D_ + d0);
            float h0 = b0 + ev.x, h1 = b1 + ev.y;
            #pragma unroll
            for (int f = 0; f < 7; ++f) {
                const float xv = xr[f];
                h0 = fmaf(xv, wc0[f], h0);
                h1 = fmaf(xv, wc1[f], h1);
            }
            *reinterpret_cast<float2*>(&hbuf[wv][m * D_ + d0]) = make_float2(h0, h1);
        }
    }
    __syncthreads();

    // h2[m] = sum_d h^2 (f32)
    if (lane < 16) {
        const float4* hr = reinterpret_cast<const float4*>(&hbuf[wv][lane * D_]);
        float s = 0.f;
        #pragma unroll
        for (int q = 0; q < 16; ++q) {
            float4 v = hr[q];
            s += v.x * v.x + v.y * v.y + v.z * v.z + v.w * v.w;
        }
        h2buf[wv][lane] = s;
    }
    __syncthreads();

    // Build A fragments for the distance GEMM (16x32 f16, contiguous k order, same as the B loads):
    // lane holds row M=row; elems 0..15 -> K = kc*32 + hi*16 + e
    v16h a0, a1;
    #pragma unroll
    for (int kc = 0; kc < 2; ++kc) {
        const float* hp = &hbuf[wv][row * D_ + kc * 32 + hi * 16];
        v16h a;
        #pragma unroll
        for (int e = 0; e < 8; ++e) { a[e] = (_Float16)hp[e]; a[8 + e] = (_Float16)hp[8 + e]; }
        if (kc == 0) a0 = a; else a1 = a;
    }

    // h2 in C/D layout: elem r <-> row r + 8*hi
    v8f h2v;
    {
        const float4* hp = reinterpret_cast<const float4*>(&h2buf[wv][hi * 8]);
        float4 u = hp[0], v = hp[1];
        h2v[0] = u.x; h2v[1] = u.y; h2v[2] = u.z; h2v[3] = u.w;
        h2v[4] = v.x; h2v[5] = v.y; h2v[6] = v.z; h2v[7] = v.w;
    }

    // -------- distance GEMM + fused exp(-dist), 16 centroid tiles --------
    v8f esum = {0.f, 0.f, 0.f, 0.f, 0.f, 0.f, 0.f, 0.f};
    #pragma unroll
    for (int ct = 0; ct < 16; ++ct) {
        const int crow = ct * 16 + row;     // this lane's centroid column N
        // B fragments: B[k][n] = cent[crow][kc*32 + hi*16 + e] (contiguous 32B)
        const v16h b0 = *reinterpret_cast<const v16h*>(&cent[crow][hi * 16]);
        const v16h b1 = *reinterpret_cast<const v16h*>(&cent[crow][32 + hi * 16]);
        v8f acc = {0.f, 0.f, 0.f, 0.f, 0.f, 0.f, 0.f, 0.f};
        acc = __builtin_amdgcn_wmma_f32_16x16x32_f16(false, a0, false, b0, (short)0, acc, false, false);
        acc = __builtin_amdgcn_wmma_f32_16x16x32_f16(false, a1, false, b1, (short)0, acc, false, false);
        const float c2c = c2s[crow];
        #pragma unroll
        for (int r = 0; r < 8; ++r) {
            const float d2   = h2v[r] + c2c - 2.f * acc[r];
            const float dist = __builtin_amdgcn_sqrtf(fmaxf(d2, 0.f));
            const float e    = __builtin_amdgcn_exp2f(-dist * LOG2E);  // exp(-dist); dist>=0 -> no overflow
            esum[r] += e;
            wbuf[wv][(r + hi * 8) * C_ + crow] = (_Float16)e;          // row-major for A reload
        }
    }

    // per-row sum of exp over all 256 centroids (reduce the 16 column-lanes)
    #pragma unroll
    for (int r = 0; r < 8; ++r) {
        float s = esum[r];
        s += __shfl_xor(s, 1, 32);
        s += __shfl_xor(s, 2, 32);
        s += __shfl_xor(s, 4, 32);
        s += __shfl_xor(s, 8, 32);
        esum[r] = s;
    }
    __syncthreads();   // make wbuf visible for fragment reloads

    // A fragments for the combine GEMM: w (unnormalized) 16x256, 8 K-chunks of 32
    v16h af[8];
    #pragma unroll
    for (int kc = 0; kc < 8; ++kc) {
        const _Float16* wp = &wbuf[wv][row * C_ + kc * 32 + hi * 16];
        const v8h lo = *reinterpret_cast<const v8h*>(wp);
        const v8h hh = *reinterpret_cast<const v8h*>(wp + 8);
        v16h a;
        #pragma unroll
        for (int e = 0; e < 8; ++e) { a[e] = lo[e]; a[8 + e] = hh[e]; }
        af[kc] = a;
    }

    // -------- combine GEMM (e @ centroids), fold W2 per d-column --------
    v8f acc2 = {0.f, 0.f, 0.f, 0.f, 0.f, 0.f, 0.f, 0.f};
    #pragma unroll
    for (int dt = 0; dt < 4; ++dt) {
        const int drow = dt * 16 + row;     // this lane's output dim column N = d
        v8f acc = {0.f, 0.f, 0.f, 0.f, 0.f, 0.f, 0.f, 0.f};
        #pragma unroll
        for (int kc = 0; kc < 8; ++kc) {
            // B[k][n] = centT[drow][kc*32 + hi*16 + e] (contiguous 32B)
            const v16h b = *reinterpret_cast<const v16h*>(&centT[drow][kc * 32 + hi * 16]);
            acc = __builtin_amdgcn_wmma_f32_16x16x32_f16(false, af[kc], false, b, (short)0, acc, false, false);
        }
        const float w2c = w2v[drow];
        #pragma unroll
        for (int r = 0; r < 8; ++r) acc2[r] += acc[r] * w2c;
    }

    // reduce over d-columns, apply deferred softmax normalization and 1/sqrt(D)=0.125
    float outv[8];
    #pragma unroll
    for (int r = 0; r < 8; ++r) {
        float s = acc2[r];
        s += __shfl_xor(s, 1, 32);
        s += __shfl_xor(s, 2, 32);
        s += __shfl_xor(s, 4, 32);
        s += __shfl_xor(s, 8, 32);
        outv[r] = s * 0.125f / esum[r] + b2s;
    }
    if (row == 0) {  // lane 0 -> rows 0..7, lane 16 -> rows 8..15
        float4* op = reinterpret_cast<float4*>(out + tbase + hi * 8);
        op[0] = make_float4(outv[0], outv[1], outv[2], outv[3]);
        op[1] = make_float4(outv[4], outv[5], outv[6], outv[7]);
    }
}

extern "C" void kernel_launch(void* const* d_in, const int* in_sizes, int n_in,
                              void* d_out, int out_size, void* d_ws, size_t ws_size,
                              hipStream_t stream) {
    (void)n_in; (void)out_size; (void)d_ws; (void)ws_size;
    const float* x    = (const float*)d_in[0];
    const float* We   = (const float*)d_in[1];
    const float* be   = (const float*)d_in[2];
    const float* emb  = (const float*)d_in[3];
    const float* cent = (const float*)d_in[4];
    const float* W2   = (const float*)d_in[5];
    const float* b2   = (const float*)d_in[6];
    float* out = (float*)d_out;

    const long tokens = (long)in_sizes[0] / F_;          // B*S = 262144
    const int  blocks = (int)(tokens / (16 * WAVES));    // 4096, exact
    hipLaunchKernelGGL(tkmeans_wmma_kernel, dim3(blocks), dim3(TPB), 0, stream,
                       x, We, be, emb, cent, W2, b2, out);
}
